// DownProj_72722386256025
// MI455X (gfx1250) — matrix operand order repaired and checked by reading, not compiled, not run on hardware
//
#include <hip/hip_runtime.h>

// Problem dims (fixed by reference)
#define M_DIM 8192
#define K_DIM 14336
#define N_DIM 4096
#define GROUP 128

// Tile config
#define BM 128
#define BN 128
#define BK 64
#define LDA 72   // padded LDS row stride (bf16 elems): 144B, conflict-free
#define LDW 72
#define NK_ITERS (K_DIM / BK)   // 224

typedef __attribute__((ext_vector_type(16))) __bf16 v16bf;
typedef __attribute__((ext_vector_type(8)))  __bf16 v8bf;
typedef __attribute__((ext_vector_type(8)))  float  v8f;
typedef __attribute__((ext_vector_type(4)))  unsigned int v4u;
typedef __attribute__((ext_vector_type(8)))  int v8i;
typedef __attribute__((ext_vector_type(4)))  int v4i;

// ---------------------------------------------------------------------------
// Pre-pass: fp32 -> bf16 conversion of activations into workspace.
// ---------------------------------------------------------------------------
__global__ __launch_bounds__(256)
void DownProj_cvt_bf16(const float* __restrict__ x, __bf16* __restrict__ y, int n8) {
  int i = blockIdx.x * blockDim.x + threadIdx.x;
  if (i >= n8) return;
  const float4* p = (const float4*)x + (size_t)i * 2;
  float4 f0 = p[0];
  float4 f1 = p[1];
  v8bf o;
  o[0] = (__bf16)f0.x; o[1] = (__bf16)f0.y; o[2] = (__bf16)f0.z; o[3] = (__bf16)f0.w;
  o[4] = (__bf16)f1.x; o[5] = (__bf16)f1.y; o[6] = (__bf16)f1.z; o[7] = (__bf16)f1.w;
  *((v8bf*)y + i) = o;
}

// ---------------------------------------------------------------------------
// TDM: DMA one 128(rows) x 64(K) bf16 tile from global [M,K] into LDS,
// padding each 128B row by 16B so the LDS row stride is 144B (= LDA elems).
// Descriptor per CDNA5 ISA (2D tensor, groups 2/3 = 0).
// This toolchain declares the 6-arg builtin form:
//   (v4u g0, v8i g1, v4i g2, v4i g3, v8i g4, i32 cpol)
// ---------------------------------------------------------------------------
static __device__ __forceinline__ void tdm_load_a_tile(const __bf16* gptr,
                                                       unsigned lds_off) {
  unsigned long long ga = (unsigned long long)gptr;
  v4u g0;
  g0[0] = 1u;                                   // count=1 (valid), user mode
  g0[1] = lds_off;                              // LDS byte address
  g0[2] = (unsigned)ga;                         // global_addr[31:0]
  g0[3] = (unsigned)((ga >> 32) & 0x01FFFFFFu)  // global_addr[56:32]
        | (2u << 30);                           // type = 2 ("image")

  v8i g1;
  g1[0] = (int)((1u << 16)      // data_size = 2 bytes
              | (1u << 20)      // pad_enable
              | (4u << 22)      // pad_interval: 32 DWORDs (128B) between pads
              | (3u << 25));    // pad_amount: 4 DWORDs (16B)
  g1[1] = (int)(((unsigned)K_DIM & 0xFFFFu) << 16);          // tensor_dim0 lo16
  g1[2] = (int)(((unsigned)K_DIM >> 16)
              | (((unsigned)M_DIM & 0xFFFFu) << 16));        // dim0 hi / dim1 lo
  g1[3] = (int)(((unsigned)M_DIM >> 16) | ((unsigned)BK << 16)); // dim1 hi / tile_dim0=64
  g1[4] = (int)(BM);                                         // tile_dim1=128, tile_dim2=0
  g1[5] = (int)(K_DIM);                                      // tensor_dim0_stride lo32
  g1[6] = 0;                                                 // stride0 hi / stride1 lo
  g1[7] = 0;                                                 // stride1 hi

  v4i z4 = {0, 0, 0, 0};
  v8i z8 = {0, 0, 0, 0, 0, 0, 0, 0};
  __builtin_amdgcn_tensor_load_to_lds(g0, g1, z4, z4, z8, 0);
}

// Dequant 8 int4 nibbles of one packed int32 -> 8 bf16 in LDS.
static __device__ __forceinline__ void dequant_store8(int w, float scale,
                                                      __bf16* dst) {
  v8bf d;
#pragma unroll
  for (int j = 0; j < 8; ++j) {
    int nib = (w >> (4 * j)) & 0xF;
    nib = (nib ^ 8) - 8;  // sign-extend 4-bit
    d[j] = (__bf16)((float)nib * scale);
  }
  *(v8bf*)dst = d;
}

// ---------------------------------------------------------------------------
// Main GEMM: y = (A_bf16 @ dequant(W)^T) * qxscale + residual + bias.
// Block 128x128, 8 wave32s (2x4), each wave 64x32 = 4x2 WMMA tiles.
// Double-buffered LDS; A tiles via TDM, W tiles via VGPR dequant.
// ---------------------------------------------------------------------------
__global__ __launch_bounds__(256)
void DownProj_gemm_wmma(const __bf16* __restrict__ A,     // [M,K] bf16 (ws)
                        const int*    __restrict__ Wq,    // [N, K/8] packed int4
                        const float*  __restrict__ Ws,    // [N, K/GROUP]
                        const float*  __restrict__ qxscale,  // [M]
                        const float*  __restrict__ residual, // [M,N]
                        const float*  __restrict__ bias,     // [N]
                        float* __restrict__ out) {           // [M,N]
  __shared__ __bf16 sA[2][BM * LDA];
  __shared__ __bf16 sW[2][BN * LDW];

  const int tid  = threadIdx.x;
  const int lane = tid & 31;
  const int wave = tid >> 5;   // 0..7
  const int wm   = wave >> 2;  // 0..1  (M direction)
  const int wn   = wave & 3;   // 0..3  (N direction)

  const int m0 = blockIdx.y * BM;
  const int n0 = blockIdx.x * BN;

  const int l16 = lane & 15;
  const int hi  = lane >> 4;

  // W staging role: each thread owns half a row (4 packed int32 = 32 K).
  const int wrow  = tid >> 1;
  const int whalf = tid & 1;

  const unsigned aoff0 = (unsigned)(unsigned long long)(void*)&sA[0][0];
  const unsigned aoff1 = (unsigned)(unsigned long long)(void*)&sA[1][0];

  v8f acc[4][2];
#pragma unroll
  for (int tm = 0; tm < 4; ++tm)
#pragma unroll
    for (int tn = 0; tn < 2; ++tn)
#pragma unroll
      for (int e = 0; e < 8; ++e) acc[tm][tn][e] = 0.0f;

  const size_t wq_row = (size_t)(n0 + wrow) * (K_DIM / 8) + whalf * 4;
  const size_t ws_row = (size_t)(n0 + wrow) * (K_DIM / GROUP);
  const __bf16* a_row = A + (size_t)m0 * K_DIM;

  // ---- prologue: TDM A tile 0, stage W tile 0 ----
  if (tid < 32) tdm_load_a_tile(a_row, aoff0);
  {
    const float sc = Ws[ws_row];
    const int4 pw = *(const int4*)(Wq + wq_row);
    __bf16* sp = &sW[0][wrow * LDW + whalf * 32];
    dequant_store8(pw.x, sc, sp + 0);
    dequant_store8(pw.y, sc, sp + 8);
    dequant_store8(pw.z, sc, sp + 16);
    dequant_store8(pw.w, sc, sp + 24);
  }
  if (tid < 32) __builtin_amdgcn_s_wait_tensorcnt(0);
  __syncthreads();

  for (int i = 0; i < NK_ITERS; ++i) {
    const int buf = i & 1;
    const int k0 = i * BK;
    const bool has_next = (i + 1) < NK_ITERS;
    const unsigned aoff_next = buf ? aoff0 : aoff1;

    // Issue next A tile DMA into the other buffer (its last readers finished
    // before the barrier that ended iteration i-1).
    if (has_next && tid < 32)
      tdm_load_a_tile(a_row + (size_t)(k0 + BK), aoff_next);

    // Preload next W packed ints + group scale into registers.
    int4 pw = {0, 0, 0, 0};
    float sc = 0.0f;
    if (has_next) {
      sc = Ws[ws_row + (unsigned)(k0 + BK) / GROUP];
      pw = *(const int4*)(Wq + wq_row + ((k0 + BK) >> 3));
      if (i + 2 < NK_ITERS)
        __builtin_prefetch(Wq + wq_row + ((k0 + 2 * BK) >> 3), 0, 0);
    }

    // ---- compute on buffer `buf`: two K=32 WMMA steps ----
    const __bf16* sAb = sA[buf];
    const __bf16* sWb = sW[buf];
#pragma unroll
    for (int kk = 0; kk < BK; kk += 32) {
      v16bf afrag[4];
#pragma unroll
      for (int tm = 0; tm < 4; ++tm) {
        const __bf16* base = &sAb[(wm * 64 + tm * 16 + l16) * LDA + kk + hi * 8];
        v8bf lo = *(const v8bf*)base;
        v8bf hh = *(const v8bf*)(base + 16);
#pragma unroll
        for (int e = 0; e < 8; ++e) { afrag[tm][e] = lo[e]; afrag[tm][e + 8] = hh[e]; }
      }
      v16bf bfrag[2];
#pragma unroll
      for (int tn = 0; tn < 2; ++tn) {
        const __bf16* base = &sWb[(wn * 32 + tn * 16 + l16) * LDW + kk + hi * 8];
        v8bf lo = *(const v8bf*)base;
        v8bf hh = *(const v8bf*)(base + 16);
#pragma unroll
        for (int e = 0; e < 8; ++e) { bfrag[tn][e] = lo[e]; bfrag[tn][e + 8] = hh[e]; }
      }
#pragma unroll
      for (int tm = 0; tm < 4; ++tm)
#pragma unroll
        for (int tn = 0; tn < 2; ++tn)
          acc[tm][tn] = __builtin_amdgcn_wmma_f32_16x16x32_bf16(
              false, afrag[tm], false, bfrag[tn], (short)0, acc[tm][tn], false, false);
    }

    // ---- stage next W tile into the other buffer; close the pipeline ----
    if (has_next) {
      __bf16* sp = &sW[buf ^ 1][wrow * LDW + whalf * 32];
      dequant_store8(pw.x, sc, sp + 0);
      dequant_store8(pw.y, sc, sp + 8);
      dequant_store8(pw.z, sc, sp + 16);
      dequant_store8(pw.w, sc, sp + 24);
      if (tid < 32) __builtin_amdgcn_s_wait_tensorcnt(0);
    }
    __syncthreads();
  }

  // ---- epilogue: per-token scale, residual, bias; fp32 stores ----
  // C/D layout: VGPR r, lane -> M = r + 8*(lane>=16), N = lane%16.
#pragma unroll
  for (int tm = 0; tm < 4; ++tm) {
#pragma unroll
    for (int tn = 0; tn < 2; ++tn) {
      const int ncol = n0 + wn * 32 + tn * 16 + l16;
      const float bv = bias[ncol];
#pragma unroll
      for (int r = 0; r < 8; ++r) {
        const int mrow = m0 + wm * 64 + tm * 16 + hi * 8 + r;
        const float v = acc[tm][tn][r] * qxscale[mrow] +
                        residual[(size_t)mrow * N_DIM + ncol] + bv;
        out[(size_t)mrow * N_DIM + ncol] = v;
      }
    }
  }
}

extern "C" void kernel_launch(void* const* d_in, const int* in_sizes, int n_in,
                              void* d_out, int out_size, void* d_ws, size_t ws_size,
                              hipStream_t stream) {
  const float* qx           = (const float*)d_in[0];
  const float* qxscale      = (const float*)d_in[1];
  const float* residual     = (const float*)d_in[2];
  const int*   weight_i4    = (const int*)d_in[3];
  const float* weight_scale = (const float*)d_in[4];
  const float* bias         = (const float*)d_in[5];
  // d_in[6] = group_size scalar (hardcoded GROUP=128)
  float* out = (float*)d_out;

  // Workspace: bf16 copy of qx, M*K*2 = 224 MiB.
  __bf16* a_bf16 = (__bf16*)d_ws;

  const int n8 = (M_DIM * K_DIM) / 8;
  DownProj_cvt_bf16<<<n8 / 256, 256, 0, stream>>>(qx, a_bf16, n8);

  dim3 grid(N_DIM / BN, M_DIM / BM);  // 32 x 64 blocks
  DownProj_gemm_wmma<<<grid, 256, 0, stream>>>(a_bf16, weight_i4, weight_scale,
                                               qxscale, residual, bias, out);
}
